// Meta_24060406792666
// MI455X (gfx1250) — matrix operand order repaired and checked by reading, not compiled, not run on hardware
//
#include <hip/hip_runtime.h>
#include <hip/hip_bf16.h>
#include <math.h>

// Problem constants (from reference): B=4, S=16, C=64, D=5, P=25, C2=66
// W1:(132,128) b1:(128) W2:(128,64) b2:(64) W3:(64,16) b3:(16) W4:(16,1) b4:(1)

typedef __attribute__((ext_vector_type(16))) _Float16 v16h;
typedef __attribute__((ext_vector_type(8)))  _Float16 v8h;
typedef __attribute__((ext_vector_type(4)))  _Float16 v4h;
typedef __attribute__((ext_vector_type(8)))  float    v8f;

// ---------------------------------------------------------------------------
// Phase 1: A'[b,s,p,h] = sum_c xf[b,s,c,p]*W1a[c,h] + b1[h]
//          Bm[b,s,p,h] = sum_c xf[b,s,c,p]*W1b[c,h]
// ---------------------------------------------------------------------------
__global__ __launch_bounds__(128) void k_embed(const float* __restrict__ x,
                                               const float* __restrict__ W1,
                                               const float* __restrict__ b1,
                                               float* __restrict__ Ap,
                                               float* __restrict__ Bm)
{
    const int row = blockIdx.x;          // (b*16+s)*25 + p
    const int h   = threadIdx.x;         // 0..127
    const int p   = row % 25;
    const int bs  = row / 25;            // b*16 + s
    const float* xb = x + (bs * 64) * 25 + p;

    float accA = 0.f, accB = 0.f;
#pragma unroll 8
    for (int c = 0; c < 64; ++c) {
        const float xv = xb[c * 25];
        accA += xv * W1[c * 128 + h];
        accB += xv * W1[(c + 66) * 128 + h];
    }
    const float ci = (float)(p / 5) * 0.2f;
    const float cj = (float)(p % 5) * 0.2f;
    accA += ci * W1[64 * 128 + h] + cj * W1[65 * 128 + h];
    accB += ci * W1[(64 + 66) * 128 + h] + cj * W1[(65 + 66) * 128 + h];

    Ap[row * 128 + h] = accA + b1[h];    // fold b1 into A'
    Bm[row * 128 + h] = accB;
}

// ---------------------------------------------------------------------------
// Phase 2 (hot): per (b,i,j): gsum[k] = sum_{r<625} relu( h1[r,:] @ W2 + b2 )[k]
// where h1[r=(p,q),h] = relu(Ap[b,j,q,h] + Bm[b,i,p,h]).  Never materialized.
//
// - Each of the 4 waves owns 10 disjoint M-tiles and ALL 4 N-tiles: one
//   composed A-fragment feeds FOUR v_wmma_f32_16x16x32_f16 ops.
// - A'/Bm rows staged in LDS as f16 (12.8 KB): compose is pure packed-f16
//   add+max straight into the fragment (no f32->f16 converts in the loop,
//   half the ds_load_b128 traffic).
// - B-fragments (4 ksteps x 4 ntiles = 128 VGPRs) loop-invariant in regs.
// - Padded rows (625..639) masked only on the single partial tile.
// ---------------------------------------------------------------------------
__global__ __launch_bounds__(128) void k_pair(const float* __restrict__ Ap,
                                              const float* __restrict__ Bm,
                                              const float* __restrict__ W2,
                                              const float* __restrict__ b2,
                                              float* __restrict__ gsum)
{
    __shared__ __align__(16) _Float16 Ash[25 * 128];
    __shared__ __align__(16) _Float16 Bsh[25 * 128];
    __shared__ float redbuf[4][64];

    const int blk = blockIdx.x;          // b*256 + i*16 + j
    const int b = blk >> 8;
    const int i = (blk >> 4) & 15;
    const int j = blk & 15;
    const int tid = threadIdx.x;

    // Cooperative load + f32->f16 convert of the 25+25 rows.
    const float4* Asrc = (const float4*)(Ap + (size_t)((b * 16 + j) * 25) * 128);
    const float4* Bsrc = (const float4*)(Bm + (size_t)((b * 16 + i) * 25) * 128);
    for (int t = tid; t < 25 * 128 / 4; t += 128) {
        const float4 va = Asrc[t];
        const float4 vb = Bsrc[t];
        ((v4h*)Ash)[t] = (v4h){(_Float16)va.x, (_Float16)va.y, (_Float16)va.z, (_Float16)va.w};
        ((v4h*)Bsh)[t] = (v4h){(_Float16)vb.x, (_Float16)vb.y, (_Float16)vb.z, (_Float16)vb.w};
    }

    const int lane = tid & 31;
    const int wav  = tid >> 5;           // wave id -> owns M-tiles wav, wav+4, ...
    const int half = lane >> 4;          // K/M half select per WMMA layouts
    const int lm   = lane & 15;

    // B fragments (loop invariant): for N-tile n, lane holds col N = lm,
    // K = 8*half + e (e<8), 16 + 8*half + (e-8) (e>=8) within k-step kb.  W2 is (128,64) row-major.
    v16h bf[4][4];
#pragma unroll
    for (int kb = 0; kb < 4; ++kb)
#pragma unroll
        for (int n = 0; n < 4; ++n)
#pragma unroll
            for (int e = 0; e < 16; ++e) {
                const int hh = kb * 32 + 8 * half + (e < 8 ? e : e + 8);    // global K index
                bf[kb][n][e] = (_Float16)W2[hh * 64 + n * 16 + lm];
            }
    float b2v[4];
#pragma unroll
    for (int n = 0; n < 4; ++n) b2v[n] = b2[n * 16 + lm];

    float partial[4] = {0.f, 0.f, 0.f, 0.f};
    const v8h hz = {(_Float16)0, (_Float16)0, (_Float16)0, (_Float16)0,
                    (_Float16)0, (_Float16)0, (_Float16)0, (_Float16)0};

    __syncthreads();                     // Ash/Bsh ready

    for (int t = 0; t < 10; ++t) {
        const int mt = wav + 4 * t;
        // A rows this lane composes: row M = lm of the tile -> r = mt*16+lm
        const int r = mt * 16 + lm;
        const int q = r % 25;
        int p = r / 25; if (p > 24) p = 24;          // clamp dead padded rows
        const v8h* arow = (const v8h*)&Ash[q * 128]; // 8-half groups
        const v8h* brow = (const v8h*)&Bsh[p * 128];

        v8f c[4];
#pragma unroll
        for (int n = 0; n < 4; ++n) c[n] = (v8f){0.f,0.f,0.f,0.f,0.f,0.f,0.f,0.f};

#pragma unroll
        for (int kb = 0; kb < 4; ++kb) {
            // A 16x32 f16 layout: lane half h -> e<8: K=8h+e ; e>=8: K=16+8h+(e-8)
            // group indices (units of 8 halves): g0 = kb*4 + half, g1 = g0 + 2
            const int g0 = kb * 4 + half;
            v8h s0 = arow[g0] + brow[g0];
            v8h s1 = arow[g0 + 2] + brow[g0 + 2];
            s0 = __builtin_elementwise_max(s0, hz);      // relu in packed f16
            s1 = __builtin_elementwise_max(s1, hz);
            const v16h af = __builtin_shufflevector(s0, s1,
                                0,1,2,3,4,5,6,7,8,9,10,11,12,13,14,15);
            // One composed A-fragment feeds 4 WMMAs (one per 16-col N-tile).
#pragma unroll
            for (int n = 0; n < 4; ++n)
                c[n] = __builtin_amdgcn_wmma_f32_16x16x32_f16(
                           false, af, false, bf[kb][n], (short)0, c[n], false, false);
        }
        // C layout: VGPR v holds row M = v + 8*half, col N = lm.
        const int rbase = mt * 16 + 8 * half;
        if (rbase + 8 <= 625) {                          // full tile: no masking
#pragma unroll
            for (int n = 0; n < 4; ++n)
#pragma unroll
                for (int v = 0; v < 8; ++v)
                    partial[n] += fmaxf(c[n][v] + b2v[n], 0.f);
        } else {                                         // only tile mt==39
#pragma unroll
            for (int n = 0; n < 4; ++n)
#pragma unroll
                for (int v = 0; v < 8; ++v) {
                    const float h2v = fmaxf(c[n][v] + b2v[n], 0.f);
                    partial[n] += (rbase + v < 625) ? h2v : 0.f;
                }
        }
    }

    // Halves hold disjoint row-sets of the same columns: combine across lane^16,
    // then sum the 4 waves' disjoint M-tile contributions through LDS.
#pragma unroll
    for (int n = 0; n < 4; ++n) {
        const float tot = partial[n] + __shfl_xor(partial[n], 16, 32);
        if (half == 0) redbuf[wav][n * 16 + lm] = tot;
    }
    __syncthreads();
    if (tid < 64) {
        const float g = redbuf[0][tid] + redbuf[1][tid] + redbuf[2][tid] + redbuf[3][tid];
        gsum[(size_t)(((b * 16 + i) * 16 + j)) * 64 + tid] = g;
    }
}

// ---------------------------------------------------------------------------
// Phase 3: per batch b (grid=4, block=256=16x16 (i,j)):
// fh = relu(gsum@W3+b3); pred = sigmoid(fh@W4+b4); block-reduce
// sum(sym^2), sum(anti^2), sum((pred-y)^2) -> partials[b*4 + {0,1,2}].
// ---------------------------------------------------------------------------
__global__ __launch_bounds__(256) void k_head(const float* __restrict__ gsum,
                                              const int* __restrict__ label,
                                              const float* __restrict__ W3,
                                              const float* __restrict__ b3,
                                              const float* __restrict__ W4,
                                              const float* __restrict__ b4,
                                              float* __restrict__ partials)
{
    __shared__ float pm[16][17];
    __shared__ float red[256];

    const int b = blockIdx.x;
    const int tid = threadIdx.x;
    const int i = tid >> 4, j = tid & 15;
    const float* g = gsum + (size_t)((b * 16 + i) * 16 + j) * 64;

    float fh[16];
#pragma unroll
    for (int t = 0; t < 16; ++t) {
        float acc = b3[t];
        for (int k = 0; k < 64; ++k) acc += g[k] * W3[k * 16 + t];
        fh[t] = fmaxf(acc, 0.f);
    }
    float z = b4[0];
#pragma unroll
    for (int t = 0; t < 16; ++t) z += fh[t] * W4[t];
    const float pred = 1.f / (1.f + expf(-z));

    pm[i][j] = pred;
    __syncthreads();
    const float pt   = pm[j][i];
    const float sym  = 0.5f * (pred + pt);
    const float anti = 0.5f * (pred - pt);
    const float y    = (label[b * 16 + i] == label[b * 16 + j]) ? 1.f : 0.f;
    const float e    = (pred - y) * (pred - y);

    float vals[3] = {sym * sym, anti * anti, e};
#pragma unroll
    for (int s = 0; s < 3; ++s) {
        red[tid] = vals[s];
        __syncthreads();
        for (int off = 128; off > 0; off >>= 1) {
            if (tid < off) red[tid] += red[tid + off];
            __syncthreads();
        }
        if (tid == 0) partials[b * 4 + s] = red[0];
        __syncthreads();
    }
}

__global__ void k_final(const float* __restrict__ partials, float* __restrict__ out)
{
    if (threadIdx.x == 0 && blockIdx.x == 0) {
        float sym_sum = 0.f, euc_sum = 0.f;
        for (int b = 0; b < 4; ++b) {
            const float sn = sqrtf(partials[b * 4 + 0]);
            const float an = sqrtf(partials[b * 4 + 1]);
            sym_sum += (sn - an) / (sn + an);
            euc_sum += partials[b * 4 + 2];
        }
        const float sym = sym_sum * 0.25f;
        const float euc = euc_sum * (1.f / 1024.f);
        out[0] = euc - 0.1f * sym;   // rn_loss
        out[1] = sym;                // sym_loss
        out[2] = euc;                // euc_loss
    }
}

// ---------------------------------------------------------------------------
// Workspace layout (floats): Ap[204800] | Bm[204800] | gsum[65536] | part[16]
// Total ~1.86 MB.
// ---------------------------------------------------------------------------
extern "C" void kernel_launch(void* const* d_in, const int* in_sizes, int n_in,
                              void* d_out, int out_size, void* d_ws, size_t ws_size,
                              hipStream_t stream) {
    const float* x     = (const float*)d_in[0];
    const int*   label = (const int*)  d_in[1];
    const float* W1    = (const float*)d_in[2];
    const float* b1    = (const float*)d_in[3];
    const float* W2    = (const float*)d_in[4];
    const float* b2    = (const float*)d_in[5];
    const float* W3    = (const float*)d_in[6];
    const float* b3    = (const float*)d_in[7];
    const float* W4    = (const float*)d_in[8];
    const float* b4    = (const float*)d_in[9];

    float* ws       = (float*)d_ws;
    float* Ap       = ws;
    float* Bm       = ws + 204800;
    float* gsum     = ws + 409600;
    float* partials = ws + 475136;
    float* out      = (float*)d_out;

    k_embed<<<1600, 128, 0, stream>>>(x, W1, b1, Ap, Bm);
    k_pair <<<1024, 128, 0, stream>>>(Ap, Bm, W2, b2, gsum);
    k_head <<<4,    256, 0, stream>>>(gsum, label, W3, b3, W4, b4, partials);
    k_final<<<1,     32, 0, stream>>>(partials, out);
}
